// ConBiMambaBlock_76888504533050
// MI455X (gfx1250) — compile-verified, matrix operand order checked by reading, not run on hardware
//
#include <hip/hip_runtime.h>
#include <math.h>

// ---------------------------------------------------------------------------
// ConBiMamba block for MI455X (gfx1250, wave32, WMMA).
// All large GEMMs run on v_wmma_f32_16x16x32_f16 (f16 in, f32 accumulate).
// Weights are repacked per launch into the wave fragment layout so B loads
// are two b128 loads per lane.  Bias/activation/residual/L-flip fused into
// the GEMM epilogue.  Scan/conv/LN are elementwise wave32 kernels.
// NOTE: every GEMM here has N % 64 == 0, so the wave's 4 sub-tiles are
// always valid -> straight-line K-loop, no control flow around WMMA.
// ---------------------------------------------------------------------------

typedef __attribute__((ext_vector_type(16))) _Float16 v16h;
typedef __attribute__((ext_vector_type(8)))  float    v8f;

#define BATCH  4096
#define SEQL   3
#define ROWS   (BATCH * SEQL)   // 12288
#define DMODEL 512
#define DINNER 1024
#define DSTATE 16
#define DTRANK 32

__device__ __forceinline__ float siluf(float x)     { return x / (1.0f + expf(-x)); }
__device__ __forceinline__ float sigmoidf_(float x) { return 1.0f / (1.0f + expf(-x)); }
__device__ __forceinline__ float softplusf_(float x){ return (x > 20.0f) ? x : log1pf(expf(x)); }

union HFrag { v16h v; uint32_t u[8]; uint4 q[2]; };

// ---------------------------------------------------------------------------
// Generic WMMA GEMM:  out = [res +] alpha * act( A(f16, MxK row-major)
//                                  x Bpacked(KxN) + bias )
// Block: 256 thr = 8 waves, each wave: 16 rows x 64 cols (4 wmma accums).
// grid.x = N/64 (N % 64 == 0), grid.y = ROWS/128.
// act: 0=none 1=silu 2=softplus.  flipL remaps output row within L=3 groups.
// ---------------------------------------------------------------------------
__global__ __launch_bounds__(256) void gemm_wmma_f16(
    const _Float16* __restrict__ A, const uint32_t* __restrict__ Bpk,
    const float* __restrict__ bias, const float* __restrict__ residual,
    float* __restrict__ out32, _Float16* __restrict__ out16,
    int K, int N, int act, float alpha, int flipL)
{
    const int lane   = threadIdx.x & 31;
    const int wv     = threadIdx.x >> 5;
    const int m0     = blockIdx.y * 128 + wv * 16;
    const int ntB    = blockIdx.x * 4;
    const int Ntiles = N >> 4;
    const int Ktiles = K >> 5;
    const int hf     = lane >> 4;          // lane half selects K sub-range
    const int mrow   = m0 + (lane & 15);

    v8f acc0 = {0,0,0,0,0,0,0,0};
    v8f acc1 = {0,0,0,0,0,0,0,0};
    v8f acc2 = {0,0,0,0,0,0,0,0};
    v8f acc3 = {0,0,0,0,0,0,0,0};

    // A fragment base: lanes 0-15 rows, V0-3: K = hf*8+0..7, V4-7: K = 16+hf*8+0..7
    const _Float16* ap = A + (size_t)mrow * K + hf * 8;
    // B fragment base: per-lane 8 consecutive u32 per (kt, nt) tile
    const uint32_t* bp = Bpk + ((size_t)(ntB * 32 + lane) << 3);
    const size_t bstride = (size_t)Ntiles << 8;   // u32 per kt step

    for (int kt = 0; kt < Ktiles; ++kt) {
        HFrag a;
        a.q[0] = *(const uint4*)(ap);
        a.q[1] = *(const uint4*)(ap + 16);
        ap += 32;
        HFrag b0, b1, b2, b3;
        b0.q[0] = *(const uint4*)(bp);       b0.q[1] = *(const uint4*)(bp + 4);
        b1.q[0] = *(const uint4*)(bp + 256); b1.q[1] = *(const uint4*)(bp + 260);
        b2.q[0] = *(const uint4*)(bp + 512); b2.q[1] = *(const uint4*)(bp + 516);
        b3.q[0] = *(const uint4*)(bp + 768); b3.q[1] = *(const uint4*)(bp + 772);
        bp += bstride;
        acc0 = __builtin_amdgcn_wmma_f32_16x16x32_f16(false, a.v, false, b0.v, (short)0, acc0, false, false);
        acc1 = __builtin_amdgcn_wmma_f32_16x16x32_f16(false, a.v, false, b1.v, (short)0, acc1, false, false);
        acc2 = __builtin_amdgcn_wmma_f32_16x16x32_f16(false, a.v, false, b2.v, (short)0, acc2, false, false);
        acc3 = __builtin_amdgcn_wmma_f32_16x16x32_f16(false, a.v, false, b3.v, (short)0, acc3, false, false);
    }

    // C/D layout: VGPR r -> M = r + 8*(lane/16), N = lane%16
    v8f accs[4] = {acc0, acc1, acc2, acc3};
#pragma unroll
    for (int s = 0; s < 4; ++s) {
        const int n  = (ntB + s) * 16 + (lane & 15);
        const float bv = bias ? bias[n] : 0.0f;
#pragma unroll
        for (int r = 0; r < 8; ++r) {
            const int m  = m0 + hf * 8 + r;
            const int mo = flipL ? ((m / 3) * 3 + 2 - (m % 3)) : m;
            float v = accs[s][r] + bv;
            if (act == 1)      v = siluf(v);
            else if (act == 2) v = softplusf_(v);
            v *= alpha;
            if (residual) v += residual[(size_t)mo * N + n];
            if (out32) out32[(size_t)mo * N + n] = v;
            if (out16) out16[(size_t)mo * N + n] = (_Float16)v;
        }
    }
}

// Repack f32 weight (KxN row-major) into per-lane B fragments (f16 pairs).
// Index decomposition mirrors the GEMM's B fetch exactly.
__global__ void repack_w(const float* __restrict__ W, uint32_t* __restrict__ out,
                         int K, int N)
{
    int idx = blockIdx.x * blockDim.x + threadIdx.x;
    int total = (K * N) >> 1;
    if (idx >= total) return;
    int v    = idx & 7;
    int lane = (idx >> 3) & 31;
    int tile = idx >> 8;                  // kt*Ntiles + nt
    int Ntiles = N >> 4;
    int nt = tile % Ntiles;
    int kt = tile / Ntiles;
    int n  = nt * 16 + (lane & 15);
    int kb = kt * 32 + ((v >> 2) << 4) + ((lane >> 4) << 3) + ((v & 3) << 1);
    union { _Float16 h[2]; uint32_t u; } p;
    p.h[0] = (_Float16)W[(size_t)kb * N + n];
    p.h[1] = (_Float16)W[(size_t)(kb + 1) * N + n];
    out[idx] = p.u;
}

// Pack [fine, coarse, motif] -> X rows (b*3 + l), l order per jnp.stack axis=1.
__global__ void pack_seq(const float* __restrict__ f, const float* __restrict__ c,
                         const float* __restrict__ m, float* __restrict__ X)
{
    int idx = blockIdx.x * blockDim.x + threadIdx.x;
    if (idx >= BATCH * DMODEL) return;
    int b = idx >> 9, d = idx & 511;
    size_t base = ((size_t)b * 3) * DMODEL + d;
    X[base]            = f[idx];
    X[base + DMODEL]   = c[idx];
    X[base + 2*DMODEL] = m[idx];
}

__global__ void neg_exp(const float* __restrict__ a, float* __restrict__ o, int n)
{
    int i = blockIdx.x * blockDim.x + threadIdx.x;
    if (i < n) o[i] = -expf(a[i]);
}

// LayerNorm over D=512; one wave per row.  Optional f16 out, flipped f16 out
// (L-reversed within each batch triple) and f32 out.
__global__ __launch_bounds__(256) void ln512(
    const float* __restrict__ x, const float* __restrict__ g, const float* __restrict__ bt,
    _Float16* __restrict__ out16, _Float16* __restrict__ out16f,
    float* __restrict__ out32, int rows)
{
    int wv = threadIdx.x >> 5, lane = threadIdx.x & 31;
    int row = blockIdx.x * 8 + wv;
    if (row >= rows) return;
    const float* xr = x + (size_t)row * DMODEL;
    float vals[16], s = 0.0f, s2 = 0.0f;
#pragma unroll
    for (int i = 0; i < 16; ++i) {
        float t = xr[lane + i * 32];
        vals[i] = t; s += t; s2 += t * t;
    }
#pragma unroll
    for (int off = 16; off >= 1; off >>= 1) {
        s  += __shfl_xor(s,  off, 32);
        s2 += __shfl_xor(s2, off, 32);
    }
    float mu  = s  * (1.0f / DMODEL);
    float var = s2 * (1.0f / DMODEL) - mu * mu;
    float rs  = rsqrtf(var + 1e-5f);
    int rowf = (row / 3) * 3 + 2 - (row % 3);
#pragma unroll
    for (int i = 0; i < 16; ++i) {
        int d = lane + i * 32;
        float y = (vals[i] - mu) * rs * g[d] + bt[d];
        if (out16)  out16 [(size_t)row  * DMODEL + d] = (_Float16)y;
        if (out16f) out16f[(size_t)rowf * DMODEL + d] = (_Float16)y;
        if (out32)  out32 [(size_t)row  * DMODEL + d] = y;
    }
}

// Mamba causal depthwise conv (K=4) + SiLU.  Input = first half of the
// in_proj output (rows x 2048 f16); output f16 xm (rows x 1024).
__global__ void mamba_conv(const _Float16* __restrict__ in2048,
                           const float* __restrict__ cw, const float* __restrict__ cb,
                           _Float16* __restrict__ xm16, int total)
{
    int idx = blockIdx.x * blockDim.x + threadIdx.x;
    if (idx >= total) return;
    int row = idx / DINNER, d = idx % DINNER;
    int b = row / 3, l = row % 3;
    float acc = cb[d];
#pragma unroll
    for (int k = 0; k < 4; ++k) {
        int t = l + k - 3;                          // causal, left pad 3
        if (t >= 0)
            acc += (float)in2048[((size_t)(b * 3 + t)) * 2048 + d] * cw[d * 4 + k];
    }
    xm16[idx] = (_Float16)siluf(acc);
}

// dbc[:, :32] -> f16 for the dt GEMM (K=32)
__global__ void extract_dt(const float* __restrict__ dbc, _Float16* __restrict__ dt16,
                           int total)
{
    int idx = blockIdx.x * blockDim.x + threadIdx.x;
    if (idx >= total) return;
    int row = idx >> 5, j = idx & 31;
    dt16[idx] = (_Float16)dbc[(size_t)row * 64 + j];
}

// Selective scan over L=3, fused with (y + D*x) * silu(z); in-place over xm16.
__global__ void mamba_scan(const float* __restrict__ dt32, const float* __restrict__ dbc,
                           const _Float16* __restrict__ z2048, const float* __restrict__ Aneg,
                           const float* __restrict__ Dp, _Float16* __restrict__ xm16,
                           int total)
{
    int idx = blockIdx.x * blockDim.x + threadIdx.x;
    if (idx >= total) return;
    int b = idx / DINNER, d = idx % DINNER;
    size_t r0 = (size_t)b * 3;
    float xm[3], zz[3], dt[3];
#pragma unroll
    for (int l = 0; l < 3; ++l) {
        xm[l] = (float)xm16[(r0 + l) * DINNER + d];
        zz[l] = (float)z2048[(r0 + l) * 2048 + DINNER + d];
        dt[l] = dt32[(r0 + l) * DINNER + d];
    }
    float h[DSTATE];
#pragma unroll
    for (int n = 0; n < DSTATE; ++n) h[n] = 0.0f;
    float Dv = Dp[d];
    for (int l = 0; l < 3; ++l) {
        float y = 0.0f;
        const float* bc = dbc + (r0 + l) * 64;
#pragma unroll
        for (int n = 0; n < DSTATE; ++n) {
            float dA = expf(dt[l] * Aneg[d * DSTATE + n]);
            h[n] = dA * h[n] + dt[l] * bc[DTRANK + n] * xm[l];
            y += h[n] * bc[DTRANK + DSTATE + n];
        }
        float val = (y + Dv * xm[l]) * siluf(zz[l]);
        xm16[(r0 + l) * DINNER + d] = (_Float16)val;
    }
}

// GLU: a * sigmoid(g) from rows x 1024 -> rows x 512
__global__ void glu_gate(const float* __restrict__ h, float* __restrict__ o, int total)
{
    int idx = blockIdx.x * blockDim.x + threadIdx.x;
    if (idx >= total) return;
    int row = idx >> 9, d = idx & 511;
    float a = h[(size_t)row * 1024 + d];
    float g = h[(size_t)row * 1024 + 512 + d];
    o[idx] = a * sigmoidf_(g);
}

// Non-causal K=31 depthwise conv over L=3 (only taps 15-l .. 17-l valid),
// + batchnorm-style affine + SiLU, output f16.
__global__ void dwconv31(const float* __restrict__ x, const float* __restrict__ w,
                         const float* __restrict__ bias, const float* __restrict__ bg,
                         const float* __restrict__ bb, _Float16* __restrict__ o, int total)
{
    int idx = blockIdx.x * blockDim.x + threadIdx.x;
    if (idx >= total) return;
    int row = idx >> 9, d = idx & 511;
    int b = row / 3, l = row % 3;
    float acc = bias[d];
#pragma unroll
    for (int j = 0; j < 3; ++j)
        acc += x[((size_t)(b * 3 + j) << 9) + d] * w[d * 31 + (15 - l + j)];
    float t = acc * bg[d] + bb[d];
    o[idx] = (_Float16)siluf(t);
}

__global__ void mean_seq(const float* __restrict__ x, float* __restrict__ o, int total)
{
    int idx = blockIdx.x * blockDim.x + threadIdx.x;
    if (idx >= total) return;
    int b = idx >> 9, d = idx & 511;
    size_t r = (size_t)b * 3 * DMODEL + d;
    o[idx] = (x[r] + x[r + DMODEL] + x[r + 2 * DMODEL]) * (1.0f / 3.0f);
}

// ---------------------------------------------------------------------------
// Host side
// ---------------------------------------------------------------------------
// Flattened input order (setup_inputs() recursive dict insertion order):
// 0 fine, 1 coarse, 2 motif,
// ffn1: 3 ln.g 4 ln.b 5 w1 6 b1 7 w2 8 b2
// ffn2: 9..14
// mamba: 15 ln.g 16 ln.b, fwd 17..25, bwd 26..34
//   (in_proj, conv_w, conv_b, x_proj, dt_w, dt_b, A_log, D, out_proj)
// conv: 35 ln.g 36 ln.b 37 pw1_w 38 pw1_b 39 dw_w 40 dw_b 41 bn_g 42 bn_b
//       43 pw2_w 44 pw2_b
// ln_out: 45 g 46 b

struct WDesc { int idx, K, N; };
static const WDesc g_wl[14] = {
    {5,  512, 2048}, {7,  2048, 512},   // ffn1 w1 w2
    {11, 512, 2048}, {13, 2048, 512},   // ffn2 w1 w2
    {17, 512, 2048}, {20, 1024, 64}, {21, 32, 1024}, {25, 1024, 512}, // fwd
    {26, 512, 2048}, {29, 1024, 64}, {30, 32, 1024}, {34, 1024, 512}, // bwd
    {37, 512, 1024}, {43, 512, 512},    // conv pw1 pw2
};

extern "C" void kernel_launch(void* const* d_in, const int* in_sizes, int n_in,
                              void* d_out, int out_size, void* d_ws, size_t ws_size,
                              hipStream_t stream)
{
    (void)in_sizes; (void)n_in; (void)out_size;
    auto f32 = [&](int i) { return (const float*)d_in[i]; };

    // ---- workspace layout ----
    size_t off = 0;
    auto take = [&](size_t bytes) -> void* {
        void* p = (char*)d_ws + off;
        off = (off + bytes + 255) & ~(size_t)255;
        return p;
    };
    size_t uofs[15]; uofs[0] = 0;
    for (int i = 0; i < 14; ++i)
        uofs[i + 1] = uofs[i] + (size_t)g_wl[i].K * g_wl[i].N / 2;
    uint32_t*  W16  = (uint32_t*)take(uofs[14] * 4);
    float*     X    = (float*)take((size_t)ROWS * DMODEL * 4);
    _Float16*  H16a = (_Float16*)take((size_t)ROWS * 2048 * 2);
    _Float16*  H16b = (_Float16*)take((size_t)ROWS * 2048 * 2);
    float*     F32a = (float*)take((size_t)ROWS * DINNER * 4);
    float*     F32b = (float*)take((size_t)ROWS * DMODEL * 4);
    float*     DBC  = (float*)take((size_t)ROWS * 64 * 4);
    _Float16*  DT16 = (_Float16*)take((size_t)ROWS * DTRANK * 2);
    float*     ANEG = (float*)take((size_t)2 * DINNER * DSTATE * 4);
    if (off > ws_size) return;   // workspace too small

    _Float16* xn16  = H16a;                               // sub0
    _Float16* xnf16 = H16a + (size_t)ROWS * DMODEL;       // sub1 (flipped LN)
    _Float16* dw16  = xnf16;                              // sub1 reuse (conv)
    _Float16* xm16  = H16a + (size_t)ROWS * DMODEL * 2;   // sub2..3 (rows x 1024)

    auto gemm = [&](const _Float16* A, const uint32_t* Bp, const float* bias,
                    const float* res, float* o32, _Float16* o16,
                    int K, int N, int act, float alpha, int flip) {
        dim3 grid(N / 64, ROWS / 128);
        gemm_wmma_f16<<<grid, dim3(256), 0, stream>>>(A, Bp, bias, res, o32, o16,
                                                      K, N, act, alpha, flip);
    };
    auto blocks = [](int total) { return dim3((total + 255) / 256); };

    // ---- setup ----
    pack_seq<<<blocks(BATCH * DMODEL), 256, 0, stream>>>(f32(0), f32(1), f32(2), X);
    for (int i = 0; i < 14; ++i) {
        int tot = g_wl[i].K * g_wl[i].N / 2;
        repack_w<<<blocks(tot), 256, 0, stream>>>(f32(g_wl[i].idx), W16 + uofs[i],
                                                  g_wl[i].K, g_wl[i].N);
    }
    neg_exp<<<blocks(DINNER * DSTATE), 256, 0, stream>>>(f32(23), ANEG, DINNER * DSTATE);
    neg_exp<<<blocks(DINNER * DSTATE), 256, 0, stream>>>(f32(32), ANEG + DINNER * DSTATE,
                                                         DINNER * DSTATE);

    // ---- x = seq + 0.5 * ffn1(seq) ----
    ln512<<<ROWS / 8, 256, 0, stream>>>(X, f32(3), f32(4), xn16, nullptr, nullptr, ROWS);
    gemm(xn16, W16 + uofs[0], f32(6), nullptr, nullptr, H16b, 512, 2048, 1, 1.0f, 0);
    gemm(H16b, W16 + uofs[1], f32(8), X, X, nullptr, 2048, 512, 0, 0.5f, 0);

    // ---- x = x + bimamba(ln(x)) ----
    ln512<<<ROWS / 8, 256, 0, stream>>>(X, f32(15), f32(16), xn16, xnf16, nullptr, ROWS);
    for (int dir = 0; dir < 2; ++dir) {
        int base = 17 + dir * 9;
        const _Float16* xn = dir ? xnf16 : xn16;
        gemm(xn, W16 + uofs[4 + dir * 4], nullptr, nullptr, nullptr, H16b,
             512, 2048, 0, 1.0f, 0);                                   // in_proj
        mamba_conv<<<blocks(ROWS * DINNER), 256, 0, stream>>>(
            H16b, f32(base + 1), f32(base + 2), xm16, ROWS * DINNER);  // conv+silu
        gemm(xm16, W16 + uofs[5 + dir * 4], nullptr, nullptr, DBC, nullptr,
             1024, 64, 0, 1.0f, 0);                                    // x_proj
        extract_dt<<<blocks(ROWS * DTRANK), 256, 0, stream>>>(DBC, DT16, ROWS * DTRANK);
        gemm(DT16, W16 + uofs[6 + dir * 4], f32(base + 5), nullptr, F32a, nullptr,
             32, 1024, 2, 1.0f, 0);                                    // dt (softplus)
        mamba_scan<<<blocks(BATCH * DINNER), 256, 0, stream>>>(
            F32a, DBC, H16b, ANEG + dir * DINNER * DSTATE, f32(base + 7),
            xm16, BATCH * DINNER);                                     // scan+gate
        gemm(xm16, W16 + uofs[7 + dir * 4], nullptr, X, X, nullptr,
             1024, 512, 0, 1.0f, dir);                                 // out_proj (+flip)
    }

    // ---- x = x + conv_module(x) ----
    ln512<<<ROWS / 8, 256, 0, stream>>>(X, f32(35), f32(36), xn16, nullptr, nullptr, ROWS);
    gemm(xn16, W16 + uofs[12], f32(38), nullptr, F32a, nullptr, 512, 1024, 0, 1.0f, 0);
    glu_gate<<<blocks(ROWS * DMODEL), 256, 0, stream>>>(F32a, F32b, ROWS * DMODEL);
    dwconv31<<<blocks(ROWS * DMODEL), 256, 0, stream>>>(
        F32b, f32(39), f32(40), f32(41), f32(42), dw16, ROWS * DMODEL);
    gemm(dw16, W16 + uofs[13], f32(44), X, X, nullptr, 512, 512, 0, 1.0f, 0);

    // ---- x = x + 0.5 * ffn2(x) ----
    ln512<<<ROWS / 8, 256, 0, stream>>>(X, f32(9), f32(10), xn16, nullptr, nullptr, ROWS);
    gemm(xn16, W16 + uofs[2], f32(12), nullptr, nullptr, H16b, 512, 2048, 1, 1.0f, 0);
    gemm(H16b, W16 + uofs[3], f32(14), X, X, nullptr, 2048, 512, 0, 0.5f, 0);

    // ---- out = ln_out(x).mean(axis=1) ----
    ln512<<<ROWS / 8, 256, 0, stream>>>(X, f32(45), f32(46), nullptr, nullptr, F32b, ROWS);
    mean_seq<<<blocks(BATCH * DMODEL), 256, 0, stream>>>(F32b, (float*)d_out,
                                                         BATCH * DMODEL);
}
